// BaseFeatureExtraction_63677185130573
// MI455X (gfx1250) — matrix operand order repaired and checked by reading, not compiled, not run on hardware
//
#include <hip/hip_runtime.h>

#define DIMC 256
#define NHEAD 16
#define BATCH 4
#define HH 128
#define WWD 128
#define HW 16384
#define EPSV 1e-5f

typedef _Float16 half16 __attribute__((ext_vector_type(16)));
typedef _Float16 half8  __attribute__((ext_vector_type(8)));
typedef _Float16 half4  __attribute__((ext_vector_type(4)));
typedef float    float8 __attribute__((ext_vector_type(8)));
typedef unsigned int uint4v __attribute__((ext_vector_type(4)));
typedef int      int8v  __attribute__((ext_vector_type(8)));
typedef int      int4v  __attribute__((ext_vector_type(4)));

// ---------------------------------------------------------------------------
// WMMA helpers (CDNA5 v_wmma_f32_16x16x32_f16, wave32)
// ---------------------------------------------------------------------------
__device__ __forceinline__ float8 wmma_f16(half16 a, half16 b, float8 c) {
  return __builtin_amdgcn_wmma_f32_16x16x32_f16(false, a, false, b, (short)0, c,
                                                false, false);
}

// A fragment 16x32 f16, row-major source (M rows contiguous in K).
// lane<16: M=lane, k = {0..7, 16..23}; lane>=16: M=lane-16, k = {8..15, 24..31}
__device__ __forceinline__ half16 frag_a_rows(const _Float16* base, int ld, int lane) {
  int ks = (lane >> 4) << 3;
  const _Float16* p = base + (lane & 15) * ld + ks;
  union { half16 v; half8 h[2]; } u;
  u.h[0] = *(const half8*)(p);
  u.h[1] = *(const half8*)(p + 16);
  return u.v;
}

// Same A fragment but gathered from a K-major (transposed) source.
__device__ __forceinline__ half16 frag_a_cols(const _Float16* base, int col, int ld,
                                              int k0, int lane) {
  int ks = k0 + ((lane >> 4) << 3);
  half16 v;
#pragma unroll
  for (int j = 0; j < 8; ++j) {
    v[j]     = base[(ks + j) * ld + col];
    v[j + 8] = base[(ks + 16 + j) * ld + col];
  }
  return v;
}

// B fragment 32x16 f16 from an N-major source (each N row contiguous in K).
// lane<16: N=lane, k = {0..7, 16..23}; lane>=16: N=lane-16, k = {8..15, 24..31}
__device__ __forceinline__ half16 frag_b_rows(const _Float16* base, int ld, int lane) {
  const _Float16* p = base + (lane & 15) * ld + ((lane >> 4) << 3);
  union { half16 v; half8 h[2]; } u;
  u.h[0] = *(const half8*)(p);
  u.h[1] = *(const half8*)(p + 16);
  return u.v;
}

// Same B fragment gathered from a K-major source B[k][n].
__device__ __forceinline__ half16 frag_b_cols(const _Float16* base, int col, int ld,
                                              int k0, int lane) {
  int ks = k0 + ((lane >> 4) << 3);
  half16 v;
#pragma unroll
  for (int j = 0; j < 8; ++j) {
    v[j]     = base[(ks + j) * ld + col];
    v[j + 8] = base[(ks + 16 + j) * ld + col];
  }
  return v;
}

// ---------------------------------------------------------------------------
// Tensor Data Mover: DMA a 128x128 f16 row-major tile from global memory into
// LDS, padding each 128-half row (64 DWORDs) with 4 DWORDs (8 halves) so the
// LDS tile lands with ld=136 halves (the layout the WMMA fragments use).
// D# fields per CDNA5 ISA.
// This toolchain exposes the 6-argument builtin:
//   (uint32x4 g0, int32x8 g1, int32x4 g2, int32x4 g3, int32x8, i32 cpol)
// ---------------------------------------------------------------------------
#if __has_builtin(__builtin_amdgcn_tensor_load_to_lds)
#define HAVE_TDM 1
__device__ __forceinline__ void tdm_load_tile_128x128_f16(const _Float16* gsrc,
                                                          _Float16* ldst) {
  unsigned long long ga = (unsigned long long)(uintptr_t)gsrc;
  unsigned int lds = (unsigned int)(uintptr_t)ldst;  // low 32 bits = LDS addr
  uint4v g0;
  g0.x = 1u;                                 // count=1 (valid user descriptor)
  g0.y = lds;                                // lds_addr
  g0.z = (unsigned int)ga;                   // global_addr[31:0]
  g0.w = (unsigned int)((ga >> 32) & 0x01FFFFFFu) | (2u << 30);  // addr[56:32]|type=2
  int8v g1;
  g1[0] = (int)((1u << 16)     // data_size = 2 bytes
              | (1u << 20)     // pad_enable
              | (5u << 22)     // pad_interval: every 64 DWORDs (code 5)
              | (3u << 25));   // pad_amount: 4 DWORDs (code 3)
  g1[1] = (int)(128u << 16);   // [47:32]=atomic_barrier_addr=0, [63:48]=tensor_dim0=128
  g1[2] = (int)(128u << 16);   // [79:64]=tensor_dim0 hi=0, [95:80]=tensor_dim1=128
  g1[3] = (int)(128u << 16);   // [111:96]=tensor_dim1 hi=0, [127:112]=tile_dim0=128
  g1[4] = 128;                 // [143:128]=tile_dim1=128, [159:144]=tile_dim2=0
  g1[5] = 128;                 // tensor_dim0_stride[31:0] = 128 elements
  g1[6] = 0;                   // stride hi / tensor_dim1_stride lo
  g1[7] = 0;
  int4v z4 = {0, 0, 0, 0};     // groups 2/3 unused (2D tensor)
  int8v z8 = {0, 0, 0, 0, 0, 0, 0, 0};
  __builtin_amdgcn_tensor_load_to_lds(g0, g1, z4, z4, z8, 0);
}
#else
#define HAVE_TDM 0
#endif

// ---------------------------------------------------------------------------
// Generic channel GEMM: out[b][o][n] = sum_c W[o][c] * X[b][c][n] (+bias,+res)
// Block: 256 threads / 8 waves, tile 64(M) x 128(N), K-step 32.
// ---------------------------------------------------------------------------
template <bool BIAS, bool RES, bool OUTH>
__global__ __launch_bounds__(256) void gemm_wmma(
    const float* __restrict__ Wm, const _Float16* __restrict__ X,
    const float* __restrict__ bias, const float* __restrict__ res,
    float* __restrict__ outF, _Float16* __restrict__ outH, int O, int C, int N) {
  __shared__ _Float16 As[64 * 40];   // [m][k], rows padded to 40 halves (80B)
  __shared__ _Float16 Bt[128 * 40];  // [n][k] transposed tile

  const int bat = blockIdx.z;
  const int nb = blockIdx.x * 128;
  const int mb = blockIdx.y * 64;
  const int t = threadIdx.x, lane = t & 31, wave = t >> 5;
  const int wm = wave & 3, wn = wave >> 2;  // 4 x 2 wave grid
  const _Float16* Xb = X + (size_t)bat * C * N;

  float8 acc[4] = {};

  for (int k0 = 0; k0 < C; k0 += 32) {
    __syncthreads();
    {  // stage weights (f32 -> f16)
      int m = t >> 2, k = (t & 3) * 8;
      const float* wp = Wm + (size_t)(mb + m) * C + k0 + k;
#pragma unroll
      for (int j = 0; j < 8; ++j) As[m * 40 + k + j] = (_Float16)wp[j];
    }
    {  // stage activations transposed: Bt[n][k] = X[k0+k][nb+n]
      int grp = t >> 5, nn = (t & 31) * 4;
#pragma unroll
      for (int kk = 0; kk < 4; ++kk) {
        int k = kk * 8 + grp;
        const _Float16* xp = Xb + (size_t)(k0 + k) * N + nb + nn;
        half4 v = *(const half4*)xp;
        if (k0 + 32 < C) __builtin_prefetch((const void*)(xp + 32 * N), 0, 1);
#pragma unroll
        for (int i = 0; i < 4; ++i) Bt[(nn + i) * 40 + k] = v[i];
      }
    }
    __syncthreads();
    half16 a = frag_a_rows(As + wm * 16 * 40, 40, lane);
#pragma unroll
    for (int nt = 0; nt < 4; ++nt) {
      half16 b = frag_b_rows(Bt + (wn * 64 + nt * 16) * 40, 40, lane);
      acc[nt] = wmma_f16(a, b, acc[nt]);
    }
  }

  const int lrow = (lane >> 4) << 3, lcol = lane & 15;
#pragma unroll
  for (int nt = 0; nt < 4; ++nt) {
    int col = nb + wn * 64 + nt * 16 + lcol;
#pragma unroll
    for (int v = 0; v < 8; ++v) {
      int row = mb + wm * 16 + lrow + v;
      float val = acc[nt][v];
      if (BIAS) val += bias[row];
      size_t oi = (size_t)bat * O * N + (size_t)row * N + col;
      if (RES) val += res[oi];
      if (OUTH) outH[oi] = (_Float16)val; else outF[oi] = val;
    }
  }
}

// ---------------------------------------------------------------------------
// Per-channel dual-axis attention: one block per (b,c), all tiles in LDS.
// ---------------------------------------------------------------------------
__device__ __forceinline__ void softmax128(const float* Sf, _Float16* Sh, int t) {
  if (t < 128) {
    const float* r = Sf + t * 132;
    float mx = r[0];
    for (int j = 1; j < 128; ++j) mx = fmaxf(mx, r[j]);
    float s = 0.f;
    _Float16* o = Sh + t * 136;
    for (int j = 0; j < 128; ++j) {
      float e = __expf(r[j] - mx);
      s += e;
      o[j] = (_Float16)e;
    }
    float inv = 1.f / s;
    for (int j = 0; j < 128; ++j) o[j] = (_Float16)((float)o[j] * inv);
  }
}

__global__ __launch_bounds__(256) void attn_kernel(const _Float16* __restrict__ qkv,
                                                   const float* __restrict__ scale,
                                                   float* __restrict__ outA) {
  extern __shared__ char smem[];
  _Float16* Qs = (_Float16*)smem;          // 128 x 136
  _Float16* Ks = Qs + 128 * 136;
  _Float16* Vs = Ks + 128 * 136;
  float* Sf = (float*)(Vs + 128 * 136);    // 128 x 132 f32 scores
  _Float16* Sh = (_Float16*)(Sf + 128 * 132);

  const int bc = blockIdx.x;
  const int b = bc >> 8, c = bc & 255;
  const int t = threadIdx.x, lane = t & 31, wave = t >> 5;
  // scale broadcasts against the HD axis in the reference -> index by c&15
  const float sc = scale[c & 15];

  const _Float16* Qg = qkv + ((size_t)(b * 3 + 0) * DIMC + c) * HW;
  const _Float16* Kg = qkv + ((size_t)(b * 3 + 1) * DIMC + c) * HW;
  const _Float16* Vg = qkv + ((size_t)(b * 3 + 2) * DIMC + c) * HW;

#if HAVE_TDM
  // Wave 0 drives three TDM transfers (global -> padded LDS tiles), then
  // drains TENSORcnt; the workgroup barrier publishes the tiles to all waves.
  if (wave == 0) {
    tdm_load_tile_128x128_f16(Qg, Qs);
    tdm_load_tile_128x128_f16(Kg, Ks);
    tdm_load_tile_128x128_f16(Vg, Vs);
    __builtin_amdgcn_s_wait_tensorcnt(0);
  }
#else
#pragma unroll
  for (int i = 0; i < 8; ++i) {
    int idx = (t + i * 256) * 8;
    int row = idx >> 7, col = idx & 127;
    *(half8*)&Qs[row * 136 + col] = *(const half8*)&Qg[idx];
    *(half8*)&Ks[row * 136 + col] = *(const half8*)&Kg[idx];
    *(half8*)&Vs[row * 136 + col] = *(const half8*)&Vg[idx];
  }
#endif
  __syncthreads();

  float* outp = outA + (size_t)bc * HW;
  const int m0 = wave * 16;
  const int lrow = (lane >> 4) << 3, lcol = lane & 15;

  // ---- horizontal pass: S[x][z] = sum_y Q[x][y] K[z][y] ----
  {
    float8 acc[8] = {};
    for (int k0 = 0; k0 < 128; k0 += 32) {
      half16 a = frag_a_rows(Qs + m0 * 136 + k0, 136, lane);
#pragma unroll
      for (int nt = 0; nt < 8; ++nt) {
        half16 bf = frag_b_rows(Ks + nt * 16 * 136 + k0, 136, lane);
        acc[nt] = wmma_f16(a, bf, acc[nt]);
      }
    }
#pragma unroll
    for (int nt = 0; nt < 8; ++nt)
#pragma unroll
      for (int v = 0; v < 8; ++v)
        Sf[(m0 + lrow + v) * 132 + nt * 16 + lcol] = acc[nt][v] * sc;
  }
  __syncthreads();
  softmax128(Sf, Sh, t);
  __syncthreads();
  // O_h[x][y] = sum_z Sh[x][z] V[z][y]
  {
    float8 acc[8] = {};
    for (int k0 = 0; k0 < 128; k0 += 32) {
      half16 a = frag_a_rows(Sh + m0 * 136 + k0, 136, lane);
#pragma unroll
      for (int nt = 0; nt < 8; ++nt) {
        half16 bf = frag_b_cols(Vs, nt * 16 + lcol, 136, k0, lane);
        acc[nt] = wmma_f16(a, bf, acc[nt]);
      }
    }
#pragma unroll
    for (int nt = 0; nt < 8; ++nt)
#pragma unroll
      for (int v = 0; v < 8; ++v)
        outp[(m0 + lrow + v) * WWD + nt * 16 + lcol] = acc[nt][v];
  }

  // ---- vertical pass: S'[y][z] = sum_x Q[x][y] K[x][z] ----
  {
    float8 acc[8] = {};
    for (int k0 = 0; k0 < 128; k0 += 32) {
      half16 a = frag_a_cols(Qs, m0 + lcol, 136, k0, lane);
#pragma unroll
      for (int nt = 0; nt < 8; ++nt) {
        half16 bf = frag_b_cols(Ks, nt * 16 + lcol, 136, k0, lane);
        acc[nt] = wmma_f16(a, bf, acc[nt]);
      }
    }
#pragma unroll
    for (int nt = 0; nt < 8; ++nt)
#pragma unroll
      for (int v = 0; v < 8; ++v)
        Sf[(m0 + lrow + v) * 132 + nt * 16 + lcol] = acc[nt][v] * sc;
  }
  __syncthreads();
  softmax128(Sf, Sh, t);
  __syncthreads();
  // out[x][y] += sum_z V[x][z] * Sh[y][z]
  {
    float8 acc[8] = {};
    for (int k0 = 0; k0 < 128; k0 += 32) {
      half16 a = frag_a_rows(Vs + m0 * 136 + k0, 136, lane);
#pragma unroll
      for (int nt = 0; nt < 8; ++nt) {
        half16 bf = frag_b_rows(Sh + nt * 16 * 136 + k0, 136, lane);
        acc[nt] = wmma_f16(a, bf, acc[nt]);
      }
    }
#pragma unroll
    for (int nt = 0; nt < 8; ++nt)
#pragma unroll
      for (int v = 0; v < 8; ++v) {
        size_t oi = (size_t)(m0 + lrow + v) * WWD + nt * 16 + lcol;
        outp[oi] += acc[nt][v];
      }
  }
}

// ---------------------------------------------------------------------------
// LayerNorm over channel dim (NCHW): thread per pixel, coalesced channel loop.
// ---------------------------------------------------------------------------
__global__ __launch_bounds__(256) void ln_kernel(const float* __restrict__ x,
                                                 const float* __restrict__ w,
                                                 const float* __restrict__ bb,
                                                 float* __restrict__ yF,
                                                 _Float16* __restrict__ yH) {
  int p = blockIdx.x * 256 + threadIdx.x;  // 0..B*HW-1
  int b = p >> 14, s = p & 16383;
  const float* xb = x + (size_t)b * DIMC * HW + s;
  float sum = 0.f, sq = 0.f;
  for (int c = 0; c < DIMC; ++c) {
    float v = xb[(size_t)c * HW];
    sum += v;
    sq += v * v;
  }
  float mu = sum * (1.f / DIMC);
  float var = sq * (1.f / DIMC) - mu * mu;
  float rstd = rsqrtf(var + EPSV);
  size_t ob = (size_t)b * DIMC * HW + s;
  for (int c = 0; c < DIMC; ++c) {
    float v = (xb[(size_t)c * HW] - mu) * rstd * w[c] + bb[c];
    if (yF) yF[ob + (size_t)c * HW] = v;
    if (yH) yH[ob + (size_t)c * HW] = (_Float16)v;
  }
}

// Global average pool per (b,c) of post-LN features.
__global__ __launch_bounds__(256) void gp_kernel(const float* __restrict__ y,
                                                 float* __restrict__ gp) {
  __shared__ float red[256];
  int bc = blockIdx.x;
  const float* p = y + (size_t)bc * HW;
  float s = 0.f;
  for (int i = threadIdx.x; i < HW; i += 256) s += p[i];
  red[threadIdx.x] = s;
  __syncthreads();
  for (int st = 128; st > 0; st >>= 1) {
    if (threadIdx.x < st) red[threadIdx.x] += red[threadIdx.x + st];
    __syncthreads();
  }
  if (threadIdx.x == 0) gp[bc] = red[0] * (1.f / HW);
}

// Tiny gate MLP: relu(gp@g1^T+b) -> softmax(.@g2^T+b) -> per-batch (cw,aw).
__global__ __launch_bounds__(256) void gate_kernel(
    const float* __restrict__ gp, const float* __restrict__ g1w,
    const float* __restrict__ g1b, const float* __restrict__ g2w,
    const float* __restrict__ g2b, float* __restrict__ gate) {
  __shared__ float hb[4 * 64];
  __shared__ float lb[4 * 2];
  int t = threadIdx.x;
  int b = t >> 6, j = t & 63;
  float a = g1b[j];
  for (int c = 0; c < DIMC; ++c) a += gp[b * DIMC + c] * g1w[j * DIMC + c];
  hb[b * 64 + j] = fmaxf(a, 0.f);
  __syncthreads();
  if (t < 8) {
    int bb2 = t >> 1, i = t & 1;
    float l = g2b[i];
    for (int k = 0; k < 64; ++k) l += hb[bb2 * 64 + k] * g2w[i * 64 + k];
    lb[bb2 * 2 + i] = l;
  }
  __syncthreads();
  if (t < 4) {
    float a0 = lb[t * 2], a1 = lb[t * 2 + 1];
    float mx = fmaxf(a0, a1);
    float e0 = __expf(a0 - mx), e1 = __expf(a1 - mx);
    float inv = 1.f / (e0 + e1);
    gate[t * 2] = e0 * inv;
    gate[t * 2 + 1] = e1 * inv;
  }
}

// Fused depthwise 3x3 + 5x5 conv (+biases) on f32 features.
__global__ __launch_bounds__(256) void conv_kernel(
    const float* __restrict__ y, const float* __restrict__ w3,
    const float* __restrict__ b3, const float* __restrict__ w5,
    const float* __restrict__ b5, float* __restrict__ out) {
  size_t idx = (size_t)blockIdx.x * 256 + threadIdx.x;
  int s = (int)(idx & 16383);
  int c = (int)((idx >> 14) & 255);
  int r = s >> 7, q = s & 127;
  const float* pl = y + (idx - (size_t)s);
  const float* W3 = w3 + c * 9;
  const float* W5 = w5 + c * 25;
  float a = b3[c] + b5[c];
  for (int dy = -2; dy <= 2; ++dy) {
    int rr = r + dy;
    if (rr < 0 || rr >= HH) continue;
    for (int dx = -2; dx <= 2; ++dx) {
      int qq = q + dx;
      if (qq < 0 || qq >= WWD) continue;
      float v = pl[rr * WWD + qq];
      a += v * W5[(dy + 2) * 5 + (dx + 2)];
      if (dy >= -1 && dy <= 1 && dx >= -1 && dx <= 1)
        a += v * W3[(dy + 1) * 3 + (dx + 1)];
    }
  }
  out[idx] = a;
}

// L2-normalize q and k rows over the spatial axis (in place, f16 storage).
__global__ __launch_bounds__(256) void qknorm_kernel(_Float16* __restrict__ qkv) {
  __shared__ float red[256];
  int r = blockIdx.x;  // 0..2047
  int tensor = r >> 10, row = r & 1023;
  int b = row >> 8, ch = row & 255;
  _Float16* p = qkv + ((size_t)(b * 3 + tensor) * DIMC + ch) * HW;
  float ss = 0.f;
  for (int i = threadIdx.x; i < HW; i += 256) {
    float v = (float)p[i];
    ss += v * v;
  }
  red[threadIdx.x] = ss;
  __syncthreads();
  for (int st = 128; st > 0; st >>= 1) {
    if (threadIdx.x < st) red[threadIdx.x] += red[threadIdx.x + st];
    __syncthreads();
  }
  float inv = 1.f / fmaxf(sqrtf(red[0]), 1e-12f);
  for (int i = threadIdx.x; i < HW; i += 256) p[i] = (_Float16)((float)p[i] * inv);
}

// mixed = cw*conv_feat + aw*attn_feat  -> f16 for the proj GEMM
__global__ __launch_bounds__(256) void mix_kernel(const float* __restrict__ cf,
                                                  const float* __restrict__ af,
                                                  const float* __restrict__ gate,
                                                  _Float16* __restrict__ out) {
  size_t idx = (size_t)blockIdx.x * 256 + threadIdx.x;
  int b = (int)(idx >> 22);
  float cw = gate[b * 2], aw = gate[b * 2 + 1];
  out[idx] = (_Float16)(cw * cf[idx] + aw * af[idx]);
}

// Grouped 3x3 conv (512ch, 2 in/out per group) fused with exact-GELU gating.
__global__ __launch_bounds__(256) void dwgelu_kernel(const _Float16* __restrict__ p,
                                                     const float* __restrict__ dw,
                                                     _Float16* __restrict__ m) {
  size_t idx = (size_t)blockIdx.x * 256 + threadIdx.x;  // over b*256*HW
  int s = (int)(idx & 16383);
  int j = (int)((idx >> 14) & 255);
  int b = (int)(idx >> 22);
  int r = s >> 7, q = s & 127;
  int g = j >> 1;
  const _Float16* base = p + (size_t)b * 512 * HW;
  const _Float16* in1 = base + (size_t)(2 * g) * HW;        // group of o=j
  const _Float16* in2 = base + (size_t)(256 + 2 * g) * HW;  // group of o=j+256
  const float* w1 = dw + (size_t)j * 18;
  const float* w2 = dw + (size_t)(j + 256) * 18;
  float acc1 = 0.f, acc2 = 0.f;
  for (int i = 0; i < 2; ++i) {
    const _Float16* p1 = in1 + (size_t)i * HW;
    const _Float16* p2 = in2 + (size_t)i * HW;
    for (int dy = -1; dy <= 1; ++dy) {
      int rr = r + dy;
      if (rr < 0 || rr >= HH) continue;
      for (int dx = -1; dx <= 1; ++dx) {
        int qq = q + dx;
        if (qq < 0 || qq >= WWD) continue;
        int off = rr * WWD + qq;
        int wi = i * 9 + (dy + 1) * 3 + (dx + 1);
        acc1 += (float)p1[off] * w1[wi];
        acc2 += (float)p2[off] * w2[wi];
      }
    }
  }
  float gl = 0.5f * acc1 * (1.f + erff(acc1 * 0.70710678118f));
  m[idx] = (_Float16)(gl * acc2);
}

// ---------------------------------------------------------------------------
extern "C" void kernel_launch(void* const* d_in, const int* in_sizes, int n_in,
                              void* d_out, int out_size, void* d_ws, size_t ws_size,
                              hipStream_t stream) {
  (void)in_sizes; (void)n_in; (void)out_size; (void)ws_size;
  const float* x      = (const float*)d_in[0];
  const float* ln1_w  = (const float*)d_in[1];
  const float* ln1_b  = (const float*)d_in[2];
  const float* conv3w = (const float*)d_in[3];
  const float* conv3b = (const float*)d_in[4];
  const float* conv5w = (const float*)d_in[5];
  const float* conv5b = (const float*)d_in[6];
  const float* qkvw   = (const float*)d_in[7];
  const float* scale  = (const float*)d_in[8];
  const float* g1w    = (const float*)d_in[9];
  const float* g1b    = (const float*)d_in[10];
  const float* g2w    = (const float*)d_in[11];
  const float* g2b    = (const float*)d_in[12];
  const float* projw  = (const float*)d_in[13];
  const float* projb  = (const float*)d_in[14];
  const float* ln2_w  = (const float*)d_in[15];
  const float* ln2_b  = (const float*)d_in[16];
  const float* pinw   = (const float*)d_in[17];
  const float* dww    = (const float*)d_in[18];
  const float* poutw  = (const float*)d_in[19];
  float* out = (float*)d_out;

  char* w = (char*)d_ws;
  const size_t BCHW = (size_t)BATCH * DIMC * HW;  // 16.7M elems
  auto take = [&](size_t bytes) {
    char* p = w;
    w += (bytes + 255) & ~(size_t)255;
    return p;
  };
  float*    yF    = (float*)take(BCHW * 4);
  _Float16* yH    = (_Float16*)take(BCHW * 2);
  float*    convf = (float*)take(BCHW * 4);
  _Float16* qkvh  = (_Float16*)take(BCHW * 3 * 2);
  float*    attnf = (float*)take(BCHW * 4);
  _Float16* mixh  = (_Float16*)take(BCHW * 2);
  float*    xnew  = (float*)take(BCHW * 4);
  _Float16* ph    = (_Float16*)take(BCHW * 2 * 2);
  _Float16* mh    = (_Float16*)take(BCHW * 2);
  float*    gp    = (float*)take(4096);
  float*    gate  = (float*)take(256);

  // 1) LN1 -> y (f32 for convs) + y (f16 for GEMMs)
  ln_kernel<<<256, 256, 0, stream>>>(x, ln1_w, ln1_b, yF, yH);
  // 2) gate path
  gp_kernel<<<BATCH * DIMC, 256, 0, stream>>>(yF, gp);
  gate_kernel<<<1, 256, 0, stream>>>(gp, g1w, g1b, g2w, g2b, gate);
  // 3) conv branch
  conv_kernel<<<65536, 256, 0, stream>>>(yF, conv3w, conv3b, conv5w, conv5b, convf);
  // 4) qkv projection (WMMA), f16 out
  gemm_wmma<false, false, true><<<dim3(128, 12, BATCH), 256, 0, stream>>>(
      qkvw, yH, nullptr, nullptr, nullptr, qkvh, 3 * DIMC, DIMC, HW);
  // 5) q/k spatial L2 normalization in place
  qknorm_kernel<<<2048, 256, 0, stream>>>(qkvh);
  // 6) dual-axis per-channel attention (WMMA + TDM staging, big dynamic LDS)
  const int attn_lds = 3 * 128 * 136 * 2 + 128 * 132 * 4 + 128 * 136 * 2;  // 206848
  (void)hipFuncSetAttribute((const void*)attn_kernel,
                            hipFuncAttributeMaxDynamicSharedMemorySize, attn_lds);
  attn_kernel<<<BATCH * DIMC, 256, attn_lds, stream>>>(qkvh, scale, attnf);
  // 7) gated mix -> f16
  mix_kernel<<<65536, 256, 0, stream>>>(convf, attnf, gate, mixh);
  // 8) proj GEMM + bias + residual -> x_new (f32)
  gemm_wmma<true, true, false><<<dim3(128, 4, BATCH), 256, 0, stream>>>(
      projw, mixh, projb, x, xnew, nullptr, DIMC, DIMC, HW);
  // 9) LN2 -> f16 only (reuse yH)
  ln_kernel<<<256, 256, 0, stream>>>(xnew, ln2_w, ln2_b, nullptr, yH);
  // 10) MLP in GEMM -> p (f16)
  gemm_wmma<false, false, true><<<dim3(128, 8, BATCH), 256, 0, stream>>>(
      pinw, yH, nullptr, nullptr, nullptr, ph, 2 * DIMC, DIMC, HW);
  // 11) grouped conv + GELU gate -> m (f16)
  dwgelu_kernel<<<65536, 256, 0, stream>>>(ph, dww, mh);
  // 12) MLP out GEMM + residual -> final output (f32)
  gemm_wmma<false, true, false><<<dim3(128, 4, BATCH), 256, 0, stream>>>(
      poutw, mh, nullptr, xnew, out, nullptr, DIMC, DIMC, HW);
}
